// GATLayer_46024869544127
// MI455X (gfx1250) — matrix operand order repaired and checked by reading, not compiled, not run on hardware
//
#include <hip/hip_runtime.h>

// ---------------- problem constants ----------------
#define NN      8192      // nodes
#define DIN     256
#define DOUTF   128
#define CHUNK   128       // K-chunk for fused attention (4 wmma k-steps)
#define PSTRIDE 136       // padded halves per sP row  (272 B = 68 dwords -> 4-bank rotate)
#define HSTRIDE 136       // padded halves per sHT row
#define LEAKY   0.01f

typedef __attribute__((ext_vector_type(16))) _Float16 v16h;
typedef __attribute__((ext_vector_type(8)))  _Float16 v8h;
typedef __attribute__((ext_vector_type(8)))  float    v8f;

// Build a v16h WMMA fragment from two contiguous 8-half (16 B) LDS runs.
static __device__ __forceinline__ v16h ld_frag(const _Float16* lo16,
                                               const _Float16* hi16) {
    v8h lo = *(const v8h*)lo16;
    v8h hi = *(const v8h*)hi16;
    return __builtin_shufflevector(lo, hi, 0, 1, 2, 3, 4, 5, 6, 7,
                                           8, 9, 10, 11, 12, 13, 14, 15);
}

// =====================================================================
// Kernel 1: h = features @ W^T + b   (WMMA f16 x f16 -> f32)
// grid = 512 blocks of 256 threads (8 wave32); block = 16 rows,
// wave w owns a 16-col N-tile; K loop = 8 steps of 32.
// Writes h as f32 [NN][DOUTF] (a1/a2 precision) and TRANSPOSED f16
// hT [DOUTF][NN] so GEMM2's B fragments are K-contiguous.
// =====================================================================
__global__ __launch_bounds__(256) void gat_h_kernel(
    const float* __restrict__ feat,   // [NN, DIN]
    const float* __restrict__ W,      // [DOUTF, DIN]
    const float* __restrict__ bias,   // [DOUTF]
    float* __restrict__ hF32,         // [NN, DOUTF]
    _Float16* __restrict__ hT)        // [DOUTF, NN]
{
    const int t    = threadIdx.x;
    const int wave = t >> 5;
    const int lane = t & 31;
    const int row0 = blockIdx.x * 16;
    const int n0   = wave * 16;
    const int col  = lane & 15;
    const int arow = lane & 15;
    const int kbA  = (lane >> 4) * 8;    // unused; A frags also use kbB (k = 16h + s)
    const int kbB  = (lane >> 4) * 16;   // A/B-frag K base, contiguous k = 16h + s

    v8f acc = {};
    for (int k0 = 0; k0 < DIN; k0 += 32) {
        v16h a, b;
#pragma unroll
        for (int v = 0; v < 8; ++v) {
            const int ka = k0 + kbB + 2 * v;
            const float* ap = feat + (size_t)(row0 + arow) * DIN + ka;
            a[2 * v]     = (_Float16)ap[0];
            a[2 * v + 1] = (_Float16)ap[1];
            const int kb = k0 + kbB + 2 * v;   // B[k][n] = W[n][k]
            const float* bp = W + (size_t)(n0 + col) * DIN + kb;
            b[2 * v]     = (_Float16)bp[0];
            b[2 * v + 1] = (_Float16)bp[1];
        }
        acc = __builtin_amdgcn_wmma_f32_16x16x32_f16(
            false, a, false, b, (short)0, acc, false, false);
    }
    const float bv = bias[n0 + col];
#pragma unroll
    for (int r = 0; r < 8; ++r) {
        const int row = r + ((lane >> 4) << 3);   // C/D layout: VGPR r -> M=r (+8)
        const float hv = acc[r] + bv;
        hF32[(size_t)(row0 + row) * DOUTF + n0 + col] = hv;
        hT[(size_t)(n0 + col) * NN + (row0 + row)]    = (_Float16)hv;
    }
}

// =====================================================================
// Kernel 2: a1[i] = h[i]·w1 + b1 ; a2[i] = h[i]·w2 + b2 ; per-block max(a2)
// =====================================================================
__global__ __launch_bounds__(256) void gat_vec_kernel(
    const float* __restrict__ hF32,
    const float* __restrict__ w1, const float* __restrict__ b1,
    const float* __restrict__ w2, const float* __restrict__ b2,
    float* __restrict__ a1, float* __restrict__ a2,
    float* __restrict__ blockMax)
{
    __shared__ float red[256];
    const int i = blockIdx.x * 256 + threadIdx.x;
    const float* hr = hF32 + (size_t)i * DOUTF;
    float s1 = 0.f, s2 = 0.f;
#pragma unroll 8
    for (int o = 0; o < DOUTF; ++o) {
        const float hv = hr[o];
        s1 += hv * w1[o];
        s2 += hv * w2[o];
    }
    const float va1 = s1 + b1[0];
    const float va2 = s2 + b2[0];
    a1[i] = va1;
    a2[i] = va2;
    red[threadIdx.x] = va2;
    __syncthreads();
    for (int off = 128; off > 0; off >>= 1) {
        if (threadIdx.x < off)
            red[threadIdx.x] = fmaxf(red[threadIdx.x], red[threadIdx.x + off]);
        __syncthreads();
    }
    if (threadIdx.x == 0) blockMax[blockIdx.x] = red[0];
}

// Kernel 3: reduce 32 block maxima -> global a2max (single wave32)
__global__ void gat_max_kernel(const float* __restrict__ blockMax,
                               float* __restrict__ a2max)
{
    float v = blockMax[threadIdx.x];
#pragma unroll
    for (int off = 16; off > 0; off >>= 1)
        v = fmaxf(v, __shfl_down(v, off, 32));
    if (threadIdx.x == 0) a2max[0] = v;
}

// =====================================================================
// Kernel 4: fused masked softmax-attention + (att @ h) via WMMA.
// grid = 512 blocks (16 rows each) x 256 threads (8 wave32).
// Per CHUNK=128 of K:
//   * issue 8 global_load_async_to_lds_b128 per thread to stage the
//     transposed h chunk (128 x 128 f16) into LDS (overlaps VALU work,
//     tracked by ASYNCcnt),
//   * compute P = exp(leaky(mask(a1+a2)) - s) into LDS f16,
//   * s_wait_asynccnt 0 + barrier,
//   * denominator partial from one b128 LDS read,
//   * 4 x v_wmma_f32_16x16x32_f16, fragments via ds_load_b128 pairs.
// Shift s_i = leaky_relu(a1[i] + max_j a2[j]) is a monotone upper bound of
// every row logit, so one pass is exact (all exps <= 1; masked -> 0).
// =====================================================================
__global__ __launch_bounds__(256) void gat_attn_kernel(
    const int* __restrict__ adj,        // [NN, NN]
    const float* __restrict__ a1,
    const float* __restrict__ a2,
    const float* __restrict__ a2maxp,
    const _Float16* __restrict__ hT,    // [DOUTF, NN]
    float* __restrict__ out)            // [NN, DOUTF]
{
    __shared__ __align__(16) _Float16 sP[16 * PSTRIDE];       // ~4.3 KB
    __shared__ __align__(16) _Float16 sHT[DOUTF * HSTRIDE];   // ~34 KB
    __shared__ float sDen[256];
    __shared__ float sDenTot[16];
    __shared__ float sA1[16];
    __shared__ float sS[16];

    const int t    = threadIdx.x;
    const int wave = t >> 5;
    const int lane = t & 31;
    const int row0 = blockIdx.x * 16;
    const int n0   = wave * 16;
    const int col  = lane & 15;
    const int arow = lane & 15;
    const int kbA  = (lane >> 4) * 8;
    const int kbB  = (lane >> 4) * 16;
    const int drow = t >> 4;
    const int dseg = t & 15;

    // LDS byte offset of sHT (generic LDS pointer: low 32 bits = LDS address)
    const unsigned ldsHT = (unsigned)(size_t)&sHT[0];

    if (t < 16) {
        const float a1v = a1[row0 + t];
        sA1[t] = a1v;
        const float m = a1v + a2maxp[0];
        sS[t] = (m > 0.f) ? m : LEAKY * m;   // leaky_relu, monotone bound
    }

    float denAcc = 0.f;
    v8f acc = {};

    for (int k0 = 0; k0 < NN; k0 += CHUNK) {
        __syncthreads();   // previous chunk's LDS readers are done

        // ---- stage hT[0..127][k0..k0+CHUNK) into LDS via async DMA (GVS):
        //      base SGPR = hT + 2*k0 (uniform), per-lane 32-bit offset.
        {
            const unsigned long long gbase =
                (unsigned long long)(size_t)hT + (unsigned long long)k0 * 2u;
#pragma unroll
            for (int s = 0; s < 8; ++s) {
                const int idx = t + 256 * s;
                const int n   = idx >> 4;          // 0..127 output channel
                const int seg = idx & 15;          // 16-byte segment in chunk
                const unsigned lds  = ldsHT + (unsigned)(n * (HSTRIDE * 2) + seg * 16);
                const unsigned voff = (unsigned)(n * (NN * 2) + seg * 16);
                asm volatile("global_load_async_to_lds_b128 %0, %1, %2"
                             :: "v"(lds), "v"(voff), "s"(gbase) : "memory");
            }
        }

        // ---- P tile: 16 x CHUNK, 8 values per thread (overlaps the DMA)
#pragma unroll
        for (int s = 0; s < (16 * CHUNK) / 256; ++s) {
            const int idx = t + 256 * s;
            const int r   = idx / CHUNK;
            const int kk  = idx % CHUNK;
            const int av  = adj[(size_t)(row0 + r) * NN + (k0 + kk)];
            const float e  = (av > 0) ? (sA1[r] + a2[k0 + kk]) : -1e16f;
            const float lr = (e > 0.f) ? e : LEAKY * e;
            sP[r * PSTRIDE + kk] = (_Float16)__expf(lr - sS[r]);
        }

        // ---- prefetch next adj chunk (global_prefetch_b8)
        if (k0 + CHUNK < NN && t < 128) {
            __builtin_prefetch(
                &adj[(size_t)(row0 + (t >> 3)) * NN + (k0 + CHUNK) + (t & 7) * 16],
                0, 0);
        }

        asm volatile("s_wait_asynccnt 0" ::: "memory");
        __syncthreads();

        // ---- denominator partial: one b128 LDS read of 8 f16 P values
        {
            v8h dv = *(const v8h*)&sP[drow * PSTRIDE + dseg * 8];
#pragma unroll
            for (int j = 0; j < 8; ++j) denAcc += (float)dv[j];
        }

        // ---- 4 WMMA k-steps over this chunk; fragments are b128 LDS loads
#pragma unroll
        for (int ks = 0; ks < CHUNK / 32; ++ks) {
            const _Float16* pa = &sP[arow * PSTRIDE + ks * 32 + kbB];
            const _Float16* pb = &sHT[(n0 + col) * HSTRIDE + ks * 32 + kbB];
            v16h a = ld_frag(pa, pa + 8);    // k runs kbB..kbB+15 contiguous
            v16h b = ld_frag(pb, pb + 8);    // k runs kbB..kbB+15 contiguous
            acc = __builtin_amdgcn_wmma_f32_16x16x32_f16(
                false, a, false, b, (short)0, acc, false, false);
        }
    }

    // ---- reduce denominators: 16 partials per row
    sDen[t] = denAcc;                 // t = drow*16 + dseg
    __syncthreads();
    if (t < 16) {
        float s = 0.f;
#pragma unroll
        for (int j = 0; j < 16; ++j) s += sDen[t * 16 + j];
        sDenTot[t] = s;
    }
    __syncthreads();

    // ---- normalize + store C tile
#pragma unroll
    for (int r = 0; r < 8; ++r) {
        const int row = r + ((lane >> 4) << 3);
        out[(size_t)(row0 + row) * DOUTF + n0 + col] = acc[r] / sDenTot[row];
    }
}

// =====================================================================
extern "C" void kernel_launch(void* const* d_in, const int* in_sizes, int n_in,
                              void* d_out, int out_size, void* d_ws, size_t ws_size,
                              hipStream_t stream) {
    const float* feat = (const float*)d_in[0];
    const int*   adj  = (const int*)d_in[1];
    const float* W    = (const float*)d_in[2];
    const float* b    = (const float*)d_in[3];
    const float* w1   = (const float*)d_in[4];
    const float* b1   = (const float*)d_in[5];
    const float* w2   = (const float*)d_in[6];
    const float* b2   = (const float*)d_in[7];
    float* out = (float*)d_out;

    char* ws = (char*)d_ws;
    float*     hF32     = (float*)(ws);                          // 4 MiB
    _Float16*  hT       = (_Float16*)(ws + (4u << 20));          // 2 MiB [DOUTF][NN]
    float*     a1       = (float*)(ws + (6u << 20));             // 32 KiB
    float*     a2       = (float*)(ws + (6u << 20) + 32768);     // 32 KiB
    float*     blockMax = (float*)(ws + (6u << 20) + 65536);     // 128 B
    float*     a2max    = (float*)(ws + (6u << 20) + 65536 + 256);

    gat_h_kernel<<<NN / 16, 256, 0, stream>>>(feat, W, b, hF32, hT);
    gat_vec_kernel<<<NN / 256, 256, 0, stream>>>(hF32, w1, b1, w2, b2,
                                                 a1, a2, blockMax);
    gat_max_kernel<<<1, 32, 0, stream>>>(blockMax, a2max);
    gat_attn_kernel<<<NN / 16, 256, 0, stream>>>(adj, a1, a2, a2max, hT, out);
}
